// S2MoELinear_8735963480503
// MI455X (gfx1250) — matrix operand order repaired and checked by reading, not compiled, not run on hardware
//
#include <hip/hip_runtime.h>
#include <hip/hip_bf16.h>

typedef __bf16 bf16_t;
typedef __attribute__((ext_vector_type(16))) __bf16 v16bf;
typedef __attribute__((ext_vector_type(8)))  __bf16 v8bf;
typedef __attribute__((ext_vector_type(8)))  float  v8f;
typedef __attribute__((ext_vector_type(4))) unsigned int u32x4;
typedef __attribute__((ext_vector_type(8))) int i32x8;
typedef __attribute__((ext_vector_type(4))) int i32x4;

#define TOK   32768   // B*S
#define DIN   1024
#define DOUT  1024
#define NE    8
#define RK    64      // GATE_K rank
#define NWID  9       // Wp + 8 experts

// ---------------- conversion kernels ----------------

__global__ __launch_bounds__(256) void cvt_x_kernel(const float* __restrict__ x,
                                                    bf16_t* __restrict__ xb) {
  size_t i = ((size_t)blockIdx.x * 256 + threadIdx.x) * 4;
  float4 v = *(const float4*)(x + i);
  xb[i + 0] = (bf16_t)v.x; xb[i + 1] = (bf16_t)v.y;
  xb[i + 2] = (bf16_t)v.z; xb[i + 3] = (bf16_t)v.w;
}

__global__ __launch_bounds__(256) void cvt_w_kernel(const float* __restrict__ Wp,
                                                    const float* __restrict__ We,
                                                    bf16_t* __restrict__ wb) {
  size_t i = ((size_t)blockIdx.x * 256 + threadIdx.x) * 4;
  const size_t wpN = (size_t)DOUT * DIN;  // 1M, multiple of 4
  const float* src = (i < wpN) ? (Wp + i) : (We + (i - wpN));
  float4 v = *(const float4*)src;
  wb[i + 0] = (bf16_t)v.x; wb[i + 1] = (bf16_t)v.y;
  wb[i + 2] = (bf16_t)v.z; wb[i + 3] = (bf16_t)v.w;
}

// VgT[e][r][k] = Vg[e][k][r]  (bf16) -> B rows become contiguous over K
__global__ __launch_bounds__(256) void cvt_vgt_kernel(const float* __restrict__ Vg,
                                                      bf16_t* __restrict__ vgt) {
  size_t id = (size_t)blockIdx.x * 256 + threadIdx.x;  // over E*RK*DIN
  int e = (int)(id >> 16);
  int rem = (int)(id & 65535);
  int r = rem >> 10;
  int k = rem & 1023;
  vgt[id] = (bf16_t)Vg[(size_t)e * DIN * RK + (size_t)k * RK + r];
}

// G[e][i][j] = sum_k Vg[e][k][i] * Vg[e][k][j]  (fp32, 8x64x64)
__global__ __launch_bounds__(64) void gmat_kernel(const float* __restrict__ Vg,
                                                  float* __restrict__ G) {
  int e = blockIdx.y, i = blockIdx.x, j = threadIdx.x;
  const float* base = Vg + (size_t)e * DIN * RK;
  float acc = 0.f;
  for (int k = 0; k < DIN; ++k)
    acc += base[(size_t)k * RK + i] * base[(size_t)k * RK + j];
  G[((size_t)e * RK + i) * RK + j] = acc;
}

// ---------------- WMMA fragment loader ----------------
__device__ inline v16bf load_frag(const bf16_t* p0, const bf16_t* p1) {
  v8bf a = *(const v8bf*)p0;
  v8bf b = *(const v8bf*)p1;
  v16bf r;
#pragma unroll
  for (int i = 0; i < 8; ++i) { r[i] = a[i]; r[8 + i] = b[i]; }
  return r;
}

// ---------------- TDM: 2-D tile (rows x 64*2B) global -> LDS ----------------
// D# per CDNA5 ISA. data_size=2B. Groups 2/3 unused (2-D).
#if __has_builtin(__builtin_amdgcn_tensor_load_to_lds)
#define HAVE_TDM 1
__device__ inline void tdm_load_2d(unsigned lds_off, const void* gaddr,
                                   unsigned tensor_d0, unsigned tensor_d1,
                                   unsigned tile_d0, unsigned tile_d1,
                                   unsigned stride0_elems) {
  unsigned long long ga = (unsigned long long)(size_t)gaddr;
  u32x4 g0;
  g0[0] = 1u;                                   // count=1, is_restore=0
  g0[1] = lds_off;                              // lds_addr (bytes)
  g0[2] = (unsigned)(ga & 0xffffffffu);         // global_addr[31:0]
  g0[3] = (unsigned)((ga >> 32) & 0x01ffffffu)  // global_addr[56:32]
          | 0x80000000u;                        // type=2 ("image")
  i32x8 g1;
  g1[0] = (int)(1u << 16);                      // wg_mask=0, data_size=1 (2B)
  g1[1] = (int)((tensor_d0 & 0xffffu) << 16);   // tensor_dim0[15:0]
  g1[2] = (int)(((tensor_d0 >> 16) & 0xffffu) | ((tensor_d1 & 0xffffu) << 16));
  g1[3] = (int)(((tensor_d1 >> 16) & 0xffffu) | ((tile_d0 & 0xffffu) << 16));
  g1[4] = (int)(tile_d1 & 0xffffu);             // tile_dim1; tile_dim2=0
  g1[5] = (int)stride0_elems;                   // tensor_dim0_stride[31:0]
  g1[6] = 0;                                    // stride0 hi / stride1 lo
  g1[7] = 0;
  i32x4 z4 = {0, 0, 0, 0};
#if __clang_major__ >= 23
  i32x8 z8 = {0, 0, 0, 0, 0, 0, 0, 0};
  __builtin_amdgcn_tensor_load_to_lds(g0, g1, z4, z4, z8, 0);
#else
  __builtin_amdgcn_tensor_load_to_lds(g0, g1, z4, z4, 0);
#endif
}
#else
#define HAVE_TDM 0
// Fallback: cooperative wave copy (32 lanes x b128)
__device__ inline void tdm_load_2d(unsigned lds_off, const void* gaddr,
                                   unsigned tensor_d0, unsigned tensor_d1,
                                   unsigned tile_d0, unsigned tile_d1,
                                   unsigned stride0_elems) {
  bf16_t* dst = (bf16_t*)(size_t)lds_off;  // not used in fallback builds
  (void)dst; (void)gaddr; (void)tensor_d0; (void)tensor_d1;
  (void)tile_d0; (void)tile_d1; (void)stride0_elems;
}
#endif

// ---------------- gate GEMM: Xe[T,512] = Xb @ VgT^T ----------------
__global__ __launch_bounds__(128) void gate_gemm_kernel(const bf16_t* __restrict__ Xb,
                                                        const bf16_t* __restrict__ VgT,
                                                        float* __restrict__ Xe) {
  int lane = threadIdx.x & 31, wv = threadIdx.x >> 5;
  int m0 = blockIdx.x * 64 + wv * 16;
  int n0 = blockIdx.y * 64;
  int halfw = lane >> 4, l16 = lane & 15;
  const bf16_t* arow = Xb + (size_t)(m0 + l16) * DIN + halfw * 8;
  v8f acc[4] = {};
  for (int k0 = 0; k0 < DIN; k0 += 32) {
    v16bf a = load_frag(arow + k0, arow + k0 + 16);
#pragma unroll
    for (int nt = 0; nt < 4; ++nt) {
      const bf16_t* brow = VgT + (size_t)(n0 + nt * 16 + l16) * DIN + k0 + halfw * 8;
      v16bf b = load_frag(brow, brow + 16);
      acc[nt] = __builtin_amdgcn_wmma_f32_16x16x32_bf16(false, a, false, b,
                                                        (short)0, acc[nt], false, false);
    }
  }
#pragma unroll
  for (int nt = 0; nt < 4; ++nt) {
    int col = n0 + nt * 16 + l16;
#pragma unroll
    for (int g = 0; g < 8; ++g) {
      int row = m0 + g + halfw * 8;
      Xe[(size_t)row * (NE * RK) + col] = acc[nt][g];
    }
  }
}

// ---------------- gate stats: one wave per token ----------------
__global__ __launch_bounds__(128) void gate_stats_kernel(const bf16_t* __restrict__ Xb,
                                                         const float* __restrict__ Xe,
                                                         const float* __restrict__ G,
                                                         float* __restrict__ Wout) {
  __shared__ float sh[4][NE * RK];
  int lane = threadIdx.x & 31, wv = threadIdx.x >> 5;
  int t = blockIdx.x * 4 + wv;

  const float* xer = Xe + (size_t)t * (NE * RK);
  for (int i = lane; i < NE * RK; i += 32) sh[wv][i] = xer[i];

  const bf16_t* xr = Xb + (size_t)t * DIN;
  float xx = 0.f;
  for (int i = lane; i < DIN; i += 32) { float v = (float)xr[i]; xx += v * v; }
#pragma unroll
  for (int off = 16; off > 0; off >>= 1) xx += __shfl_xor(xx, off, 32);

  float res[NE];
#pragma unroll
  for (int e = 0; e < NE; ++e) {
    const float* Ge = G + (size_t)e * RK * RK;
    const float* xee = &sh[wv][e * RK];
    float s = 0.f, q = 0.f;
#pragma unroll
    for (int h = 0; h < 2; ++h) {
      int i = lane + h * 32;
      float xi = xee[i];
      s += xi * xi;
      float y = 0.f;
#pragma unroll 8
      for (int j = 0; j < RK; ++j) y += Ge[i * RK + j] * xee[j];
      q += xi * y;
    }
#pragma unroll
    for (int off = 16; off > 0; off >>= 1) {
      s += __shfl_xor(s, off, 32);
      q += __shfl_xor(q, off, 32);
    }
    res[e] = sqrtf(fmaxf(xx - 2.f * s + q, 0.f));
  }

  float mn = res[0];
#pragma unroll
  for (int e = 1; e < NE; ++e) mn = fminf(mn, res[e]);
  float p[NE], sum = 0.f;
#pragma unroll
  for (int e = 0; e < NE; ++e) { p[e] = __expf(mn - res[e]); sum += p[e]; }
  float inv = 1.f / sum;
  float rw[NE];
#pragma unroll
  for (int e = 0; e < NE; ++e) rw[e] = p[e] * inv;

  float b1 = -1.f, b2 = -1.f; int i1 = -1, i2 = -1;
#pragma unroll
  for (int e = 0; e < NE; ++e) {
    if (rw[e] > b1)      { b2 = b1; i2 = i1; b1 = rw[e]; i1 = e; }
    else if (rw[e] > b2) { b2 = rw[e]; i2 = e; }
  }
  const float thr = 1.0f / NE;
  float fw[NE], ssum = 0.f;
#pragma unroll
  for (int e = 0; e < NE; ++e) {
    bool keep = (rw[e] > thr) && (e == i1 || e == i2);
    fw[e] = keep ? rw[e] : 0.f;
    ssum += fw[e];
  }
  float nrm = (ssum > 0.f) ? 1.f / ssum : 0.f;
  if (lane == 0) {
#pragma unroll
    for (int e = 0; e < NE; ++e) Wout[(size_t)t * NE + e] = fw[e] * nrm;
  }
}

// ---------------- fused 9-matrix MoE GEMM, TDM double-buffered ----------------
__global__ __launch_bounds__(128) void moe_gemm_kernel(const bf16_t* __restrict__ Xb,
                                                       const bf16_t* __restrict__ Wb,
                                                       const float* __restrict__ bp,
                                                       const float* __restrict__ be,
                                                       const float* __restrict__ Wgate,
                                                       float* __restrict__ Out) {
  __shared__ __align__(16) bf16_t shA[2][64 * 64];   // 2 x 8KB
  __shared__ __align__(16) bf16_t shB[2][64 * 64];   // 2 x 8KB
  __shared__ float sw[64 * NE];

  int tid = threadIdx.x;
  int m0b = blockIdx.x * 64;
  int n0  = blockIdx.y * 64;
  for (int i = tid; i < 64 * NE; i += 128) sw[i] = Wgate[(size_t)m0b * NE + i];

  int lane = tid & 31, wv = tid >> 5;
  int halfw = lane >> 4, l16 = lane & 15;

  const int NCHUNK = NWID * (DIN / 64);  // 144 chunks of K=64

  auto issue = [&](int c, int buf) {
#if HAVE_TDM
    int wid = c >> 4, kc = c & 15;
    int k0 = kc * 64;
    const bf16_t* ga = Xb + (size_t)m0b * DIN + k0;
    const bf16_t* gb = Wb + (size_t)wid * DOUT * DIN + (size_t)n0 * DIN + k0;
    tdm_load_2d((unsigned)(size_t)(void*)&shA[buf][0], ga, DIN, TOK, 64, 64, DIN);
    tdm_load_2d((unsigned)(size_t)(void*)&shB[buf][0], gb, DIN, DOUT, 64, 64, DIN);
#else
    int wid = c >> 4, kc = c & 15;
    int k0 = kc * 64;
    const bf16_t* ga = Xb + (size_t)m0b * DIN + k0;
    const bf16_t* gb = Wb + (size_t)wid * DOUT * DIN + (size_t)n0 * DIN + k0;
    for (int r = lane; r < 64; r += 32) {
      *(v8bf*)&shA[buf][r * 64]      = *(const v8bf*)(ga + (size_t)r * DIN);
      *(v8bf*)&shA[buf][r * 64 + 8]  = *(const v8bf*)(ga + (size_t)r * DIN + 8);
      *(v8bf*)&shA[buf][r * 64 + 16] = *(const v8bf*)(ga + (size_t)r * DIN + 16);
      *(v8bf*)&shA[buf][r * 64 + 24] = *(const v8bf*)(ga + (size_t)r * DIN + 24);
      *(v8bf*)&shA[buf][r * 64 + 32] = *(const v8bf*)(ga + (size_t)r * DIN + 32);
      *(v8bf*)&shA[buf][r * 64 + 40] = *(const v8bf*)(ga + (size_t)r * DIN + 40);
      *(v8bf*)&shA[buf][r * 64 + 48] = *(const v8bf*)(ga + (size_t)r * DIN + 48);
      *(v8bf*)&shA[buf][r * 64 + 56] = *(const v8bf*)(ga + (size_t)r * DIN + 56);
      *(v8bf*)&shB[buf][r * 64]      = *(const v8bf*)(gb + (size_t)r * DIN);
      *(v8bf*)&shB[buf][r * 64 + 8]  = *(const v8bf*)(gb + (size_t)r * DIN + 8);
      *(v8bf*)&shB[buf][r * 64 + 16] = *(const v8bf*)(gb + (size_t)r * DIN + 16);
      *(v8bf*)&shB[buf][r * 64 + 24] = *(const v8bf*)(gb + (size_t)r * DIN + 24);
      *(v8bf*)&shB[buf][r * 64 + 32] = *(const v8bf*)(gb + (size_t)r * DIN + 32);
      *(v8bf*)&shB[buf][r * 64 + 40] = *(const v8bf*)(gb + (size_t)r * DIN + 40);
      *(v8bf*)&shB[buf][r * 64 + 48] = *(const v8bf*)(gb + (size_t)r * DIN + 48);
      *(v8bf*)&shB[buf][r * 64 + 56] = *(const v8bf*)(gb + (size_t)r * DIN + 56);
    }
#endif
  };

  v8f facc[4] = {};
  v8f tacc[4] = {};

  if (wv == 0) { issue(0, 0); issue(1, 1); }

  for (int c = 0; c < NCHUNK; ++c) {
    int buf = c & 1;
    int wid = c >> 4, kc = c & 15;
#if HAVE_TDM
    if (wv == 0) {
      if (c + 1 < NCHUNK) __builtin_amdgcn_s_wait_tensorcnt(2);  // chunk c done
      else                __builtin_amdgcn_s_wait_tensorcnt(0);
    }
#endif
    __syncthreads();  // buffer c ready for everyone

    if (kc == 0) {
#pragma unroll
      for (int nt = 0; nt < 4; ++nt)
#pragma unroll
        for (int g = 0; g < 8; ++g) tacc[nt][g] = 0.f;
    }

#pragma unroll
    for (int km = 0; km < 2; ++km) {
      int k0m = km * 32;
      const bf16_t* ap = &shA[buf][(wv * 16 + l16) * 64 + k0m + halfw * 8];
      v16bf a = load_frag(ap, ap + 16);
#pragma unroll
      for (int nt = 0; nt < 4; ++nt) {
        const bf16_t* bptr = &shB[buf][(nt * 16 + l16) * 64 + k0m + halfw * 8];
        v16bf b = load_frag(bptr, bptr + 16);
        tacc[nt] = __builtin_amdgcn_wmma_f32_16x16x32_bf16(false, a, false, b,
                                                           (short)0, tacc[nt], false, false);
      }
    }

    if (kc == 15) {  // fold this matrix into final accumulator
      if (wid == 0) {
#pragma unroll
        for (int nt = 0; nt < 4; ++nt)
#pragma unroll
          for (int g = 0; g < 8; ++g) facc[nt][g] += tacc[nt][g];
      } else {
#pragma unroll
        for (int g = 0; g < 8; ++g) {
          float ws = sw[(wv * 16 + g + halfw * 8) * NE + (wid - 1)];
#pragma unroll
          for (int nt = 0; nt < 4; ++nt) facc[nt][g] += ws * tacc[nt][g];
        }
      }
    }

    __syncthreads();  // all reads of buffer c done before it is overwritten
    if (wv == 0 && c + 2 < NCHUNK) issue(c + 2, buf);
  }

  // epilogue: + bp[col] + sum_e w[row][e]*be[e][col]
#pragma unroll
  for (int nt = 0; nt < 4; ++nt) {
    int col = n0 + nt * 16 + l16;
    float bpv = bp[col];
    float bev[NE];
#pragma unroll
    for (int e = 0; e < NE; ++e) bev[e] = be[(size_t)e * DOUT + col];
#pragma unroll
    for (int g = 0; g < 8; ++g) {
      int lrow = wv * 16 + g + halfw * 8;
      float v = facc[nt][g] + bpv;
#pragma unroll
      for (int e = 0; e < NE; ++e) v += sw[lrow * NE + e] * bev[e];
      Out[(size_t)(m0b + lrow) * DOUT + col] = v;
    }
  }
}

// ---------------- launch ----------------
extern "C" void kernel_launch(void* const* d_in, const int* in_sizes, int n_in,
                              void* d_out, int out_size, void* d_ws, size_t ws_size,
                              hipStream_t stream) {
  const float* x  = (const float*)d_in[0];
  const float* Wp = (const float*)d_in[1];
  const float* bp = (const float*)d_in[2];
  const float* We = (const float*)d_in[3];
  const float* be = (const float*)d_in[4];
  const float* Vg = (const float*)d_in[5];
  float* out = (float*)d_out;

  char* ws = (char*)d_ws;
  constexpr size_t o_xb  = 0;
  constexpr size_t o_wb  = o_xb  + (size_t)TOK * DIN * 2;          // 64 MB
  constexpr size_t o_vgt = o_wb  + (size_t)NWID * DOUT * DIN * 2;  // +18 MB
  constexpr size_t o_g   = o_vgt + (size_t)NE * RK * DIN * 2;      // +1 MB
  constexpr size_t o_xe  = o_g   + (size_t)NE * RK * RK * 4;       // +128 KB
  constexpr size_t o_w   = o_xe  + (size_t)TOK * NE * RK * 4;      // +64 MB
  bf16_t* xb  = (bf16_t*)(ws + o_xb);
  bf16_t* wb  = (bf16_t*)(ws + o_wb);
  bf16_t* vgt = (bf16_t*)(ws + o_vgt);
  float*  G   = (float*)(ws + o_g);
  float*  xe  = (float*)(ws + o_xe);
  float*  wg  = (float*)(ws + o_w);

  cvt_x_kernel<<<(TOK * (size_t)DIN) / (256 * 4), 256, 0, stream>>>(x, xb);
  cvt_w_kernel<<<((size_t)NWID * DOUT * DIN) / (256 * 4), 256, 0, stream>>>(Wp, We, wb);
  cvt_vgt_kernel<<<((size_t)NE * RK * DIN) / 256, 256, 0, stream>>>(Vg, vgt);
  gmat_kernel<<<dim3(RK, NE), RK, 0, stream>>>(Vg, G);
  gate_gemm_kernel<<<dim3(TOK / 64, (NE * RK) / 64), 128, 0, stream>>>(xb, vgt, xe);
  gate_stats_kernel<<<TOK / 4, 128, 0, stream>>>(xb, xe, G, wg);
  moe_gemm_kernel<<<dim3(TOK / 64, DOUT / 64), 128, 0, stream>>>(xb, wb, bp, be, wg, out);
}
